// LatticeLSTM_15685220565616
// MI455X (gfx1250) — compile-verified, matrix operand order checked by reading, not run on hardware
//
#include <hip/hip_runtime.h>
#include <hip/hip_bf16.h>

// ---------------------------------------------------------------------------
// Lattice LSTM on MI455X (gfx1250, wave32, WMMA).
//
// Phase 1 (parallel): pre-swizzle activations + weights into WMMA fragment-
//   major bf16, then dense v_wmma GEMMs for all input-side terms.
// Phase 2 (persistent, 16 WGs = 16 WGPs): sequential t-loop; all 8 batch rows
//   packed into one M=16 WMMA tile; gate columns partitioned across WGs; the
//   full recurrent weight slice (224 KB) lives in LDS (dynamic alloc, 320 KB
//   WGP budget), staged once via GLOBAL_LOAD_ASYNC_TO_LDS_B128; two device-
//   scope barriers per step exchange h / cw through L2.
// ---------------------------------------------------------------------------

typedef __attribute__((ext_vector_type(16))) __bf16 v16bf;
typedef __attribute__((ext_vector_type(8)))  float  v8f;

#define B_   8
#define T_   1024
#define DC_  128
#define H_   512
#define H3_  1536
#define NWG  16            // persistent workgroups == H/32 chunks

static __device__ __forceinline__ unsigned short f2bf(float f) {
  union { float f; unsigned u; } c; c.f = f;
  unsigned u = c.u;
  unsigned r = (u + 0x7FFFu + ((u >> 16) & 1u)) >> 16;   // RNE
  return (unsigned short)r;
}
static __device__ __forceinline__ float bf2f(unsigned short h) {
  union { unsigned u; float f; } c; c.u = ((unsigned)h) << 16; return c.f;
}
static __device__ __forceinline__ float sigm(float x) { return 1.0f / (1.0f + __expf(-x)); }
// 16-bit A/B fragment K-mapping (ISA 7.12.2): lane-group grp = lane>>4.
static __device__ __forceinline__ int kmap(int e, int grp) {
  return ((e >> 3) << 4) + (grp << 3) + (e & 7);
}

// Async DMA: global -> LDS, 16 bytes per lane, tracked by ASYNCcnt.
// VDST VGPR carries the LDS byte offset (low 32 bits of a generic LDS ptr).
static __device__ __forceinline__ void async_cp16(unsigned lds_off, const void* gsrc) {
  asm volatile("global_load_async_to_lds_b128 %0, %1, off"
               :: "v"(lds_off), "v"((unsigned long long)(uintptr_t)gsrc)
               : "memory");
}
static __device__ __forceinline__ void wait_async0() {
  asm volatile("s_wait_asynccnt 0" ::: "memory");
}

// ------------------------------------------------- fragment-major swizzles
// A: f32 row-major [M][128] -> bf16 [M/16][kt=4][lane=32][e=16]
__global__ void swz_a(const float* __restrict__ src, unsigned short* __restrict__ dst,
                      int M) {
  int total = M * DC_;
  for (int idx = blockIdx.x * blockDim.x + threadIdx.x; idx < total;
       idx += gridDim.x * blockDim.x) {
    int e = idx & 15, lane = (idx >> 4) & 31, kt = (idx >> 9) & 3, mt = idx >> 11;
    int row = mt * 16 + (lane & 15);
    int k = kt * 32 + kmap(e, lane >> 4);
    dst[idx] = f2bf(src[(size_t)row * DC_ + k]);
  }
}

// A with sense-table gather: rows indexed by word_ids.
__global__ void swz_a_gather(const float* __restrict__ sense, const int* __restrict__ wids,
                             unsigned short* __restrict__ dst, int M) {
  int total = M * DC_;
  for (int idx = blockIdx.x * blockDim.x + threadIdx.x; idx < total;
       idx += gridDim.x * blockDim.x) {
    int e = idx & 15, lane = (idx >> 4) & 31, kt = (idx >> 9) & 3, mt = idx >> 11;
    int row = mt * 16 + (lane & 15);
    int k = kt * 32 + kmap(e, lane >> 4);
    dst[idx] = f2bf(sense[(size_t)wids[row] * DC_ + k]);
  }
}

// B: f32 row-major W[128][N] -> bf16 [N/16][kt=4][lane=32][e=16]
__global__ void swz_w(const float* __restrict__ src, unsigned short* __restrict__ dst,
                      int N) {
  int total = N * DC_;
  for (int idx = blockIdx.x * blockDim.x + threadIdx.x; idx < total;
       idx += gridDim.x * blockDim.x) {
    int e = idx & 15, lane = (idx >> 4) & 31, kt = (idx >> 9) & 3, nt = idx >> 11;
    int col = nt * 16 + (lane & 15);
    int k = kt * 32 + kmap(e, lane >> 4);
    dst[idx] = f2bf(src[(size_t)k * N + col]);
  }
}

// Recurrent weights W[512][Nfull] -> per-WG slices [wg][kt=16][n][lane][e].
// Slice column cc -> global col (cc/32)*512 + wg*32 + cc%32 (gate-split).
__global__ void stage_fm(const float* __restrict__ W, unsigned short* __restrict__ dst,
                         int Nfull, int ntiles) {
  int total = NWG * 16 * ntiles * 512;
  for (int idx = blockIdx.x * blockDim.x + threadIdx.x; idx < total;
       idx += gridDim.x * blockDim.x) {
    int e    = idx & 15;
    int lane = (idx >> 4) & 31;
    int rest = idx >> 9;                 // (wg*16 + kt)*ntiles + n
    int n    = rest % ntiles;
    int kt   = (rest / ntiles) & 15;
    int wg   = rest / (ntiles * 16);
    int row  = kt * 32 + kmap(e, lane >> 4);
    int col  = (n >> 1) * H_ + wg * 32 + (n & 1) * 16 + (lane & 15);
    dst[idx] = f2bf(W[(size_t)row * Nfull + col]);
  }
}

__global__ void init_state(unsigned short* __restrict__ hstate,
                           unsigned short* __restrict__ cwring, int* __restrict__ sync) {
  int gid = blockIdx.x * blockDim.x + threadIdx.x;
  for (int i = gid; i < 2 * 8192; i += gridDim.x * blockDim.x) hstate[i] = 0;
  for (int i = gid; i < 4 * 8192; i += gridDim.x * blockDim.x) cwring[i] = 0;
  if (gid < 8) sync[gid] = 0;
}

// ------------------------------------------------------- phase-1 WMMA GEMM
// C[M x N] = A @ W + bias, operands fragment-major. Block: 16 rows x 256
// cols; 8 waves x 2 N-tiles each (A fragment reused across both).
__global__ __launch_bounds__(256) void wmma_gemm_fm(
    const unsigned short* __restrict__ A, const unsigned short* __restrict__ W,
    const float* __restrict__ bias, float* __restrict__ C, int N) {
  const int wave = threadIdx.x >> 5, lane = threadIdx.x & 31;
  const int grp = lane >> 4, m = lane & 15;
  const int mt = blockIdx.x;
  const int nt0 = blockIdx.y * 16 + wave * 2;
  const unsigned short* ap  = A + (size_t)mt * 2048 + lane * 16;
  const unsigned short* wp0 = W + (size_t)nt0 * 2048 + lane * 16;
  const unsigned short* wp1 = wp0 + 2048;
  v8f acc0 = {}, acc1 = {};
#pragma unroll
  for (int kt = 0; kt < 4; ++kt) {
    v16bf a  = *(const v16bf*)(ap  + kt * 512);
    v16bf w0 = *(const v16bf*)(wp0 + kt * 512);
    v16bf w1 = *(const v16bf*)(wp1 + kt * 512);
    acc0 = __builtin_amdgcn_wmma_f32_16x16x32_bf16(false, a, false, w0, (short)0, acc0,
                                                   false, false);
    acc1 = __builtin_amdgcn_wmma_f32_16x16x32_bf16(false, a, false, w1, (short)0, acc1,
                                                   false, false);
  }
  const int col0 = nt0 * 16 + m;
  const float bv0 = bias[col0], bv1 = bias[col0 + 16];
#pragma unroll
  for (int v = 0; v < 8; ++v) {
    const size_t r = (size_t)(mt * 16 + grp * 8 + v) * N;
    C[r + col0] = acc0[v] + bv0;
    C[r + col0 + 16] = acc1[v] + bv1;
  }
}

// ----------------------------------------------------- device-scope barrier
static __device__ __forceinline__ void grid_sync(int* cnt, int* gen) {
  __syncthreads();
  if (threadIdx.x == 0) {
    __threadfence();
    int g = __hip_atomic_load(gen, __ATOMIC_ACQUIRE, __HIP_MEMORY_SCOPE_AGENT);
    int arrived = __hip_atomic_fetch_add(cnt, 1, __ATOMIC_ACQ_REL, __HIP_MEMORY_SCOPE_AGENT);
    if (arrived == NWG - 1) {
      __hip_atomic_store(cnt, 0, __ATOMIC_RELAXED, __HIP_MEMORY_SCOPE_AGENT);
      __hip_atomic_fetch_add(gen, 1, __ATOMIC_RELEASE, __HIP_MEMORY_SCOPE_AGENT);
    } else {
      while (__hip_atomic_load(gen, __ATOMIC_ACQUIRE, __HIP_MEMORY_SCOPE_AGENT) == g)
        __builtin_amdgcn_s_sleep(4);
    }
    __threadfence();
  }
  __syncthreads();
}

// ----------------------------------------------------- phase-2 persistent
// Dynamic LDS holds the full recurrent weight slice (224 KB < 320 KB WGP):
//   [0      .. 49152) sWhh   [49152 .. 98304) sWWhh   [98304 ..114688) sAW
// hstate: [2][8192] halves, A-fragment-major [16 tiles][32][16], rows=batch.
// cwring: [4][8192] halves, rows m = 8*word + batch. Masks analytic:
// m0 = t>=1, m1 = t>=3 (lengths 2 and 4, set-once/consume-once ring).
__global__ __launch_bounds__(256, 1) void lattice_recurrent(
    const float* __restrict__ gates_x, const float* __restrict__ ax,
    const float* __restrict__ wgx,
    const unsigned short* __restrict__ WhhFM, const unsigned short* __restrict__ WWhhFM,
    const unsigned short* __restrict__ AWhhFM,
    unsigned short* __restrict__ hstate, unsigned short* __restrict__ cwring,
    float* __restrict__ out, int* __restrict__ sync) {
  extern __shared__ __align__(16) unsigned short smem[];
  unsigned short* sWhh  = smem;             // 16*6*512 halves
  unsigned short* sWWhh = smem + 49152;     // 16*6*512 halves
  unsigned short* sAW   = smem + 98304;     // 16*2*512 halves
  __shared__ float gbuf[16][96];   // gate tile (rows 0..7 live)
  __shared__ float abuf[16][32];   // alpha tile, rows m = 8*word + batch
  __shared__ float cst[8][32];     // persistent c chunk

  const int wg = blockIdx.x;
  const int tid = threadIdx.x;
  const int wave = tid >> 5, lane = tid & 31;
  const int grp = lane >> 4, m16 = lane & 15;
  const int bb = wave;             // elementwise: batch = wave, cc = lane

  // One-shot async DMA of this WG's weight slices into LDS (ASYNCcnt path).
  // All trip counts are multiples of 256 -> EXEC stays all-ones.
  {
    const char* gW = (const char*)(WhhFM + (size_t)wg * 49152);
    const unsigned ldsW = (unsigned)(uintptr_t)sWhh;
    for (int i = tid; i < 6144; i += 256) async_cp16(ldsW + i * 16, gW + (size_t)i * 16);
    const char* gWW = (const char*)(WWhhFM + (size_t)wg * 49152);
    const unsigned ldsWW = (unsigned)(uintptr_t)sWWhh;
    for (int i = tid; i < 6144; i += 256) async_cp16(ldsWW + i * 16, gWW + (size_t)i * 16);
    const char* gA = (const char*)(AWhhFM + (size_t)wg * 16384);
    const unsigned ldsA = (unsigned)(uintptr_t)sAW;
    for (int i = tid; i < 2048; i += 256) async_cp16(ldsA + i * 16, gA + (size_t)i * 16);
    wait_async0();
  }
  cst[wave][lane] = 0.0f;
  __syncthreads();

  for (int t = 0; t < T_; ++t) {
    const unsigned short* hprev = hstate + ((t & 1) ^ 1) * 8192;

    // ---- G1: char gates (waves 0..5) + alpha = cins@aw_hh (waves 6..7) ----
    if (wave < 6) {
      v8f acc = {};
#pragma unroll
      for (int kt = 0; kt < 16; ++kt) {
        v16bf a = *(const v16bf*)(hprev + (kt * 32 + lane) * 16);
        v16bf w = *(const v16bf*)(sWhh + ((kt * 6 + wave) * 32 + lane) * 16);
        acc = __builtin_amdgcn_wmma_f32_16x16x32_bf16(false, a, false, w, (short)0, acc,
                                                      false, false);
      }
#pragma unroll
      for (int v = 0; v < 8; ++v) gbuf[grp * 8 + v][wave * 16 + m16] = acc[v];
    } else {
      const int n = wave - 6;
      const unsigned short* cin = cwring + (size_t)(t & 3) * 8192;
      v8f acc = {};
#pragma unroll
      for (int kt = 0; kt < 16; ++kt) {
        v16bf a = *(const v16bf*)(cin + (kt * 32 + lane) * 16);
        v16bf w = *(const v16bf*)(sAW + ((kt * 2 + n) * 32 + lane) * 16);
        acc = __builtin_amdgcn_wmma_f32_16x16x32_bf16(false, a, false, w, (short)0, acc,
                                                      false, false);
      }
#pragma unroll
      for (int v = 0; v < 8; ++v) abuf[grp * 8 + v][n * 16 + m16] = acc[v];
    }
    __syncthreads();

    // ---- elementwise char cell: thread (bb, cc) owns h-dim wg*32+cc ----
    {
      const int cc = lane;
      const int gc = wg * 32 + cc;
      const float* gx = gates_x + (size_t)(bb * T_ + t) * H3_;
      float iv = sigm(gx[gc] + gbuf[bb][cc]);
      float ov = sigm(gx[H_ + gc] + gbuf[bb][32 + cc]);
      float gv = tanhf(gx[2 * H_ + gc] + gbuf[bb][64 + cc]);
      const int em = ((cc >> 4) << 3) | (cc & 7);   // inverse fragment map
      const int cg = (cc >> 3) & 1;
      float c1;
      if (t >= 1) {                                  // pmask[r].any()
        float axv = ax[(size_t)(bb * T_ + t) * H_ + gc];
        float ei = __expf(iv);
        float num = ei * gv, den = ei;
        const unsigned short* cin = cwring + (size_t)(t & 3) * 8192;
        {
          float a0 = sigm(axv + abuf[bb][cc]);
          float e0 = __expf(a0);
          num += e0 * bf2f(cin[(wg * 32 + (bb + 16 * cg)) * 16 + em]);
          den += e0;
        }
        if (t >= 3) {
          float a1 = sigm(axv + abuf[8 + bb][cc]);
          float e1 = __expf(a1);
          num += e1 * bf2f(cin[(wg * 32 + (8 + bb + 16 * cg)) * 16 + em]);
          den += e1;
        }
        c1 = num / den;
      } else {
        c1 = (1.0f - iv) * cst[bb][cc] + iv * gv;    // f = 1 - i path
      }
      float h1 = ov * tanhf(c1);
      cst[bb][cc] = c1;
      out[(size_t)(bb * T_ + t) * H_ + gc] = h1;                          // hs
      out[(size_t)B_ * T_ * H_ + (size_t)(bb * T_ + t) * H_ + gc] = c1;   // cs
      hstate[(t & 1) * 8192 + (wg * 32 + (bb + 16 * cg)) * 16 + em] = f2bf(h1);
      if (t + 1 < T_)
        __builtin_prefetch(gates_x + (size_t)(bb * T_ + t + 1) * H3_ + gc, 0, 0);
    }
    grid_sync(sync, sync + 1);   // S1: h1(t) published device-wide

    // ---- G2: word gates = h1 @ ww_hh (waves 0..5) ----
    const unsigned short* hcur = hstate + (t & 1) * 8192;
    if (wave < 6) {
      v8f acc = {};
#pragma unroll
      for (int kt = 0; kt < 16; ++kt) {
        v16bf a = *(const v16bf*)(hcur + (kt * 32 + lane) * 16);
        v16bf w = *(const v16bf*)(sWWhh + ((kt * 6 + wave) * 32 + lane) * 16);
        acc = __builtin_amdgcn_wmma_f32_16x16x32_bf16(false, a, false, w, (short)0, acc,
                                                      false, false);
      }
#pragma unroll
      for (int v = 0; v < 8; ++v) gbuf[grp * 8 + v][wave * 16 + m16] = acc[v];
    }
    __syncthreads();

    // ---- elementwise word cell, write cw into ring ----
    {
      const int cc = lane, gc = wg * 32 + cc;
      const float c1 = cst[bb][cc];
      const int em = ((cc >> 4) << 3) | (cc & 7);
      const int cg = (cc >> 3) & 1;
#pragma unroll
      for (int j = 0; j < 2; ++j) {
        const int len = j ? 4 : 2;
        if (t + len - 1 < T_) {
          const float* wx = wgx + ((size_t)(bb * T_ + t) * 2 + j) * H3_;
          float fv = sigm(wx[gc] + gbuf[bb][cc]);
          float iw = sigm(wx[H_ + gc] + gbuf[bb][32 + cc]);
          float gw = tanhf(wx[2 * H_ + gc] + gbuf[bb][64 + cc]);
          float cw = fv * c1 + iw * gw;
          cwring[(size_t)((t + len - 1) & 3) * 8192 +
                 (wg * 32 + (8 * j + bb + 16 * cg)) * 16 + em] = f2bf(cw);
        }
      }
    }
    grid_sync(sync, sync + 1);   // S2: cw ring published
  }
}

// --------------------------------------------------------------- launcher
extern "C" void kernel_launch(void* const* d_in, const int* in_sizes, int n_in,
                              void* d_out, int out_size, void* d_ws, size_t ws_size,
                              hipStream_t stream) {
  (void)in_sizes; (void)n_in; (void)out_size; (void)ws_size;
  const float* char_emb = (const float*)d_in[0];
  const int*   word_ids = (const int*)d_in[1];
  const float* sense    = (const float*)d_in[2];
  const float* w_ih     = (const float*)d_in[3];
  const float* w_hh     = (const float*)d_in[4];
  const float* bias     = (const float*)d_in[5];
  const float* aw_ih    = (const float*)d_in[6];
  const float* aw_hh    = (const float*)d_in[7];
  const float* ab       = (const float*)d_in[8];
  const float* ww_ih    = (const float*)d_in[9];
  const float* ww_hh    = (const float*)d_in[10];
  const float* wb       = (const float*)d_in[11];
  float* out = (float*)d_out;

  char* ws = (char*)d_ws;
  size_t off = 0;
  auto alloc = [&](size_t bytes) -> char* {
    char* p = ws + off;
    off = (off + bytes + 255) & ~(size_t)255;
    return p;
  };
  unsigned short* XcFM   = (unsigned short*)alloc((size_t)B_ * T_ * DC_ * 2);
  unsigned short* XwFM   = (unsigned short*)alloc((size_t)B_ * T_ * 2 * DC_ * 2);
  unsigned short* WihFM  = (unsigned short*)alloc((size_t)DC_ * H3_ * 2);
  unsigned short* AWihFM = (unsigned short*)alloc((size_t)DC_ * H_ * 2);
  unsigned short* WWihFM = (unsigned short*)alloc((size_t)DC_ * H3_ * 2);
  unsigned short* WhhFM  = (unsigned short*)alloc((size_t)H_ * H3_ * 2);
  unsigned short* WWhhFM = (unsigned short*)alloc((size_t)H_ * H3_ * 2);
  unsigned short* AWhhFM = (unsigned short*)alloc((size_t)H_ * H_ * 2);
  float* gatesx = (float*)alloc((size_t)B_ * T_ * H3_ * 4);
  float* axbuf  = (float*)alloc((size_t)B_ * T_ * H_ * 4);
  float* wgxbuf = (float*)alloc((size_t)B_ * T_ * 2 * H3_ * 4);
  unsigned short* hstate = (unsigned short*)alloc(2 * 8192 * 2);
  unsigned short* cwring = (unsigned short*)alloc(4 * 8192 * 2);
  int* sync = (int*)alloc(256);

  // fragment-major conversions (activations, input-side + recurrent weights)
  swz_a<<<512, 256, 0, stream>>>(char_emb, XcFM, B_ * T_);
  swz_a_gather<<<512, 256, 0, stream>>>(sense, word_ids, XwFM, B_ * T_ * 2);
  swz_w<<<128, 256, 0, stream>>>(w_ih, WihFM, H3_);
  swz_w<<<64, 256, 0, stream>>>(aw_ih, AWihFM, H_);
  swz_w<<<128, 256, 0, stream>>>(ww_ih, WWihFM, H3_);
  stage_fm<<<256, 256, 0, stream>>>(w_hh, WhhFM, H3_, 6);
  stage_fm<<<256, 256, 0, stream>>>(ww_hh, WWhhFM, H3_, 6);
  stage_fm<<<128, 256, 0, stream>>>(aw_hh, AWhhFM, H_, 2);
  init_state<<<64, 256, 0, stream>>>(hstate, cwring, sync);

  // phase-1 input-side GEMMs (bias folded in); blocks cover 16x256 of C
  wmma_gemm_fm<<<dim3(B_ * T_ / 16, H3_ / 256), 256, 0, stream>>>(XcFM, WihFM, bias,
                                                                  gatesx, H3_);
  wmma_gemm_fm<<<dim3(B_ * T_ / 16, H_ / 256), 256, 0, stream>>>(XcFM, AWihFM, ab,
                                                                 axbuf, H_);
  wmma_gemm_fm<<<dim3(B_ * T_ * 2 / 16, H3_ / 256), 256, 0, stream>>>(XwFM, WWihFM, wb,
                                                                      wgxbuf, H3_);

  // phase-2 persistent recurrence: 224 KB dynamic LDS per WG (weight slices)
  lattice_recurrent<<<NWG, 256, 229376, stream>>>(gatesx, axbuf, wgxbuf, WhhFM, WWhhFM,
                                                  AWhhFM, hstate, cwring, out, sync);
}
